// attention_propagantion_41918880809176
// MI455X (gfx1250) — matrix operand order repaired and checked by reading, not compiled, not run on hardware
//
#include <hip/hip_runtime.h>
#include <math.h>

#define B_  8
#define C_  256
#define H_  4
#define D_  64
#define N_  2048
#define M_  2048
#define C2_ 512
#define EPS_ 1e-5f

typedef __attribute__((ext_vector_type(16))) __bf16 bf16x16;
typedef __attribute__((ext_vector_type(8)))  float  f32x8;
typedef __attribute__((ext_vector_type(4)))  int    i32x4;

// ---------------------------------------------------------------------------
// Async global->LDS staging (CDNA5 GLOBAL_LOAD_ASYNC_TO_LDS_B128, ASYNCcnt).
// Signature (from hipcc diagnostic): (v4i addrspace(1)*, v4i addrspace(3)*, Imm, Imm)
// Guarded so the file compiles on toolchains without the builtins.
// ---------------------------------------------------------------------------
#if defined(__has_builtin)
#if __has_builtin(__builtin_amdgcn_global_load_async_to_lds_b128) && \
    __has_builtin(__builtin_amdgcn_s_wait_asynccnt)
#define HAS_ASYNC_LDS 1
#endif
#endif
#ifndef HAS_ASYNC_LDS
#define HAS_ASYNC_LDS 0
#endif

typedef __attribute__((address_space(1))) i32x4 g_i32x4;
typedef __attribute__((address_space(3))) i32x4 l_i32x4;

__device__ __forceinline__ void stage16(const void* g, void* l) {
#if HAS_ASYNC_LDS
  __builtin_amdgcn_global_load_async_to_lds_b128((g_i32x4*)g, (l_i32x4*)l, 0, 0);
#else
  *(uint4*)l = *(const uint4*)g;
#endif
}
__device__ __forceinline__ void stage_join() {
#if HAS_ASYNC_LDS
  __builtin_amdgcn_s_wait_asynccnt(0);
#endif
}

union BF2 { unsigned int u; __bf16 h[2]; };

// pack two consecutive bf16 (4-byte aligned) into fragment VGPR v
__device__ __forceinline__ void frag_set2(bf16x16& f, int v, const __bf16* p) {
  BF2 t; t.u = *(const unsigned int*)p;
  f[2 * v] = t.h[0]; f[2 * v + 1] = t.h[1];
}

// A-operand 16x32 bf16 fragment; source stored [row][k] k-contiguous, leading dim ldk.
// Layout per ISA 7.12.2: lanes 0-15 row r hold K 0..7 & 16..23; lanes 16-31 hold K 8..15 & 24..31.
__device__ __forceinline__ bf16x16 load_frag_a(const __bf16* base, int ldk) {
  const int lane = threadIdx.x & 31;
  const int r = lane & 15, hf = lane >> 4;
  const __bf16* rp = base + r * ldk;
  bf16x16 f;
#pragma unroll
  for (int v = 0; v < 8; ++v) {
    int k0 = (v < 4) ? (hf * 8 + 2 * v) : (16 + hf * 8 + 2 * (v - 4));
    frag_set2(f, v, rp + k0);
  }
  return f;
}

// B-operand 32x16 bf16 fragment; source stored [col][k] k-contiguous.
// Lanes 0-15 hold K 0..7 & 16..23; lanes 16-31 hold K 8..15 & 24..31 (2 per VGPR).
__device__ __forceinline__ bf16x16 load_frag_b(const __bf16* base, int ldk) {
  const int lane = threadIdx.x & 31;
  const int c = lane & 15, hf = lane >> 4;
  const __bf16* cp = base + c * ldk;
  bf16x16 f;
#pragma unroll
  for (int v = 0; v < 8; ++v) {
    int k0 = (v < 4) ? (hf * 8 + 2 * v) : (16 + hf * 8 + 2 * (v - 4));
    frag_set2(f, v, cp + k0);
  }
  return f;
}

__device__ __forceinline__ f32x8 wmma_bf16(bf16x16 a, bf16x16 b, f32x8 c) {
  return __builtin_amdgcn_wmma_f32_16x16x32_bf16(false, a, false, b, (short)0, c, false, false);
}

// ---------------------------------------------------------------------------
// Generic channel GEMM: Out[b] = W(Cout x Cin) * X[b](Cin x Ncols) + bias
// X K-dim can come from two stacked sources (k < ksplit -> X0, else X1).
// Epilogues: 0 plain f32 [b][c][n]; 1 bf16 per-head transposed [b][h][n][d];
//            2 bf16 [b][h][d][m] * weight_v; 3 BN+ReLU f32; 4 residual f32.
// (fp32 -> bf16 conversion happens while staging, so staging stays on VGPRs.)
// ---------------------------------------------------------------------------
template<int EPI>
__global__ __launch_bounds__(256)
void gemm_wmma_kernel(const float* __restrict__ W, const float* __restrict__ bias,
                      const float* __restrict__ X0, long sX0,
                      const float* __restrict__ X1, long sX1, int ksplit,
                      int Cout, int Cin, int Ncols,
                      float* __restrict__ outF, __bf16* __restrict__ outB,
                      const float* __restrict__ wv,
                      const float* __restrict__ bng, const float* __restrict__ bnb,
                      const float* __restrict__ bnm, const float* __restrict__ bnv,
                      const float* __restrict__ resid)
{
  __shared__ __bf16 Alds[64 * 32];   // weights tile  [row][k]
  __shared__ __bf16 Blds[128 * 32];  // act tile transposed [n][k]

  const int tid   = threadIdx.x;
  const int wave  = tid >> 5;
  const int row_w = wave & 3;   // 4 row groups x 16 rows
  const int col_w = wave >> 2;  // 2 col groups x 64 cols
  const int b     = blockIdx.z;
  const int tile_r = blockIdx.y * 64;
  const int tile_c = blockIdx.x * 128;

  f32x8 acc[4] = {};

  for (int k0 = 0; k0 < Cin; k0 += 32) {
#pragma unroll
    for (int i = 0; i < 8; ++i) {              // 64x32 weights
      int idx = tid + i * 256;
      int r = idx >> 5, kk = idx & 31;
      Alds[idx] = (__bf16)W[(long)(tile_r + r) * Cin + k0 + kk];
    }
#pragma unroll
    for (int i = 0; i < 16; ++i) {             // 32x128 acts -> [n][k]
      int idx = tid + i * 256;
      int kk = idx >> 7, n = idx & 127;
      int gk = k0 + kk;
      const float* Xs = (gk < ksplit)
          ? (X0 + (long)b * sX0 + (long)gk * Ncols)
          : (X1 + (long)b * sX1 + (long)(gk - ksplit) * Ncols);
      Blds[n * 32 + kk] = (__bf16)Xs[tile_c + n];
    }
    __syncthreads();

    bf16x16 af = load_frag_a(Alds + row_w * 16 * 32, 32);
#pragma unroll
    for (int ct = 0; ct < 4; ++ct) {
      bf16x16 bfr = load_frag_b(Blds + (col_w * 64 + ct * 16) * 32, 32);
      acc[ct] = wmma_bf16(af, bfr, acc[ct]);
    }
    __syncthreads();
  }

  const int lane = tid & 31;
  const int lc = lane & 15, hf = lane >> 4;
#pragma unroll
  for (int ct = 0; ct < 4; ++ct) {
#pragma unroll
    for (int v = 0; v < 8; ++v) {
      int c = tile_r + row_w * 16 + v + hf * 8;        // output channel
      int n = tile_c + col_w * 64 + ct * 16 + lc;      // output column
      float val = acc[ct][v] + bias[c];
      if constexpr (EPI == 0) {
        outF[((long)b * Cout + c) * Ncols + n] = val;
      } else if constexpr (EPI == 1) {                 // Q/K: [b][h][n][d] bf16
        int h = c >> 6, d = c & 63;
        outB[(((long)b * H_ + h) * Ncols + n) * D_ + d] = (__bf16)val;
      } else if constexpr (EPI == 2) {                 // V: *wv, [b][h][d][m] bf16
        val *= wv[(long)b * Ncols + n];
        int h = c >> 6, d = c & 63;
        outB[(((long)b * H_ + h) * D_ + d) * Ncols + n] = (__bf16)val;
      } else if constexpr (EPI == 3) {                 // BN + ReLU
        float inv = bng[c] * rsqrtf(bnv[c] + EPS_);
        val = fmaxf(val * inv + (bnb[c] - bnm[c] * inv), 0.0f);
        outF[((long)b * Cout + c) * Ncols + n] = val;
      } else {                                         // residual add
        val += resid[((long)b * Cout + c) * Ncols + n];
        outF[((long)b * Cout + c) * Ncols + n] = val;
      }
    }
  }
}

// ---------------------------------------------------------------------------
// Flash attention: per (b,h), 8 waves x 16 rows of N = 128 rows per block.
// S = softmax(Q^T K / 8) streamed over M in 64-column tiles; O += P * V^T.
// K/V tiles staged via GLOBAL_LOAD_ASYNC_TO_LDS_B128 when available.
// ---------------------------------------------------------------------------
__global__ __launch_bounds__(256)
void attention_wmma_kernel(const __bf16* __restrict__ qb, const __bf16* __restrict__ kb,
                           const __bf16* __restrict__ vb, float* __restrict__ attn)
{
  __shared__ __bf16 Kt[64 * 64];       // [m][d]
  __shared__ __bf16 Vt[64 * 64];       // [d][m]
  __shared__ __bf16 Pl[8 * 16 * 64];   // per-wave P tile [16 n][64 m]

  const int tid  = threadIdx.x;
  const int wave = tid >> 5;
  const int lane = tid & 31;
  const int lc = lane & 15, hf = lane >> 4;
  const int b = blockIdx.z, h = blockIdx.y;
  const int n0 = blockIdx.x * 128 + wave * 16;

  const __bf16* qbase = qb + (((long)b * H_ + h) * N_ + n0) * D_;
  const __bf16* kbase = kb + ((long)b * H_ + h) * (long)M_ * D_;
  const __bf16* vbase = vb + ((long)b * H_ + h) * (long)D_ * M_;

  // Q fragments resident for the whole kernel (rows n, k = d)
  bf16x16 qa0 = load_frag_a(qbase, D_);
  bf16x16 qa1 = load_frag_a(qbase + 32, D_);

  f32x8 oacc[4] = {};
  float mrow[8], lrow[8];
#pragma unroll
  for (int v = 0; v < 8; ++v) { mrow[v] = -INFINITY; lrow[v] = 0.f; }

  for (int m0 = 0; m0 < M_; m0 += 64) {
    __syncthreads();                    // prior-iteration Kt/Vt reads complete
    // stage K [64 m][64 d] (contiguous) and V^T [64 d][64 m] via async-to-LDS
    {
      const __bf16* ksrc = kbase + (long)m0 * D_;
#pragma unroll
      for (int i = 0; i < 2; ++i) {     // 2 x b128 per thread
        int idx = tid + i * 256;
        stage16(ksrc + idx * 8, Kt + idx * 8);
      }
#pragma unroll
      for (int i = 0; i < 2; ++i) {     // 8 x b128 per d-row
        int idx = tid + i * 256;
        int d = idx >> 3, cv = idx & 7;
        stage16(vbase + (long)d * M_ + m0 + cv * 8, Vt + d * 64 + cv * 8);
      }
      if (m0 + 64 < M_)                 // global_prefetch_b8 for next K tile
        __builtin_prefetch(kbase + (long)(m0 + 64) * D_ + (long)tid * 8, 0, 1);
      stage_join();                     // s_wait_asynccnt 0 (own wave's issues)
    }
    __syncthreads();                    // cross-wave visibility

    // S tiles: rows n (A = Q), cols m (B = K), k = d (2 chunks of 32)
    f32x8 s[4];
#pragma unroll
    for (int mt = 0; mt < 4; ++mt) {
      f32x8 z = {};
      z = wmma_bf16(qa0, load_frag_b(Kt + mt * 16 * 64, 64), z);
      z = wmma_bf16(qa1, load_frag_b(Kt + mt * 16 * 64 + 32, 64), z);
      s[mt] = z;
    }

    // online softmax; C-frag rows live in one 16-lane half, cols over lanes
    __bf16* Pw = Pl + wave * 16 * 64;
#pragma unroll
    for (int v = 0; v < 8; ++v) {
      float mx = -INFINITY;
#pragma unroll
      for (int mt = 0; mt < 4; ++mt) { s[mt][v] *= 0.125f; mx = fmaxf(mx, s[mt][v]); }
#pragma unroll
      for (int off = 1; off < 16; off <<= 1) mx = fmaxf(mx, __shfl_xor(mx, off, 16));
      float mnew = fmaxf(mrow[v], mx);
      float corr = __expf(mrow[v] - mnew);
      mrow[v] = mnew;
      float rs = 0.f;
#pragma unroll
      for (int mt = 0; mt < 4; ++mt) {
        float p = __expf(s[mt][v] - mnew);
        rs += p;
        Pw[(v + hf * 8) * 64 + mt * 16 + lc] = (__bf16)p;   // D-layout -> A-layout via LDS
      }
#pragma unroll
      for (int off = 1; off < 16; off <<= 1) rs += __shfl_xor(rs, off, 16);
      lrow[v] = lrow[v] * corr + rs;
#pragma unroll
      for (int dt = 0; dt < 4; ++dt) oacc[dt][v] *= corr;
    }

    // O += P * V^T   (A = P rows n / k = m; B = V^T cols d / k = m)
    bf16x16 pa0 = load_frag_a(Pw, 64);
    bf16x16 pa1 = load_frag_a(Pw + 32, 64);
#pragma unroll
    for (int dt = 0; dt < 4; ++dt) {
      oacc[dt] = wmma_bf16(pa0, load_frag_b(Vt + dt * 16 * 64, 64), oacc[dt]);
      oacc[dt] = wmma_bf16(pa1, load_frag_b(Vt + dt * 16 * 64 + 32, 64), oacc[dt]);
    }
  }

  // normalize and store attn output as f32 [b][c][n] (feeds Wm GEMM)
#pragma unroll
  for (int dt = 0; dt < 4; ++dt) {
#pragma unroll
    for (int v = 0; v < 8; ++v) {
      int c = h * 64 + dt * 16 + lc;
      int n = n0 + v + hf * 8;
      attn[((long)b * C_ + c) * N_ + n] = oacc[dt][v] / lrow[v];
    }
  }
}

// ---------------------------------------------------------------------------
extern "C" void kernel_launch(void* const* d_in, const int* in_sizes, int n_in,
                              void* d_out, int out_size, void* d_ws, size_t ws_size,
                              hipStream_t stream) {
  (void)in_sizes; (void)n_in; (void)out_size; (void)ws_size;
  const float* desc1 = (const float*)d_in[0];
  const float* desc2 = (const float*)d_in[1];
  const float* wv    = (const float*)d_in[2];
  const float* Wq = (const float*)d_in[3];  const float* bq  = (const float*)d_in[4];
  const float* Wk = (const float*)d_in[5];  const float* bk  = (const float*)d_in[6];
  const float* Wvm = (const float*)d_in[7]; const float* bv  = (const float*)d_in[8];
  const float* Wm = (const float*)d_in[9];  const float* bm  = (const float*)d_in[10];
  const float* Wc1 = (const float*)d_in[11]; const float* bc1 = (const float*)d_in[12];
  const float* bng = (const float*)d_in[13]; const float* bnb = (const float*)d_in[14];
  const float* bnm = (const float*)d_in[15]; const float* bnv = (const float*)d_in[16];
  const float* Wc2 = (const float*)d_in[17]; const float* bc2 = (const float*)d_in[18];
  float* out = (float*)d_out;

  char* ws = (char*)d_ws;
  size_t off = 0;
  auto wsalloc = [&](size_t bytes) -> void* {
    void* p = ws + off;
    off = (off + bytes + 255) & ~(size_t)255;
    return p;
  };
  __bf16* qbf = (__bf16*)wsalloc((size_t)B_ * H_ * N_ * D_ * 2);  // [b][h][n][d]
  __bf16* kbf = (__bf16*)wsalloc((size_t)B_ * H_ * M_ * D_ * 2);  // [b][h][m][d]
  __bf16* vbf = (__bf16*)wsalloc((size_t)B_ * H_ * D_ * M_ * 2);  // [b][h][d][m]
  float*  attn = (float*)wsalloc((size_t)B_ * C_ * N_ * 4);       // [b][c][n]
  float*  am   = (float*)wsalloc((size_t)B_ * C_ * N_ * 4);       // Wm output
  float*  hb   = (float*)wsalloc((size_t)B_ * C2_ * N_ * 4);      // BN/ReLU output

  const dim3 blk(256);

  // Q = Wq*desc1+bq -> bf16 [b][h][n][d]
  gemm_wmma_kernel<1><<<dim3(N_/128, C_/64, B_), blk, 0, stream>>>(
      Wq, bq, desc1, (long)C_ * N_, nullptr, 0, C_, C_, C_, N_,
      nullptr, qbf, nullptr, nullptr, nullptr, nullptr, nullptr, nullptr);
  // K = Wk*desc2+bk -> bf16 [b][h][m][d]
  gemm_wmma_kernel<1><<<dim3(M_/128, C_/64, B_), blk, 0, stream>>>(
      Wk, bk, desc2, (long)C_ * M_, nullptr, 0, C_, C_, C_, M_,
      nullptr, kbf, nullptr, nullptr, nullptr, nullptr, nullptr, nullptr);
  // V = (Wv*desc2+bv)*weight_v -> bf16 [b][h][d][m]
  gemm_wmma_kernel<2><<<dim3(M_/128, C_/64, B_), blk, 0, stream>>>(
      Wvm, bv, desc2, (long)C_ * M_, nullptr, 0, C_, C_, C_, M_,
      nullptr, vbf, wv, nullptr, nullptr, nullptr, nullptr, nullptr);

  // flash attention core
  attention_wmma_kernel<<<dim3(N_/128, H_, B_), blk, 0, stream>>>(qbf, kbf, vbf, attn);

  // am = Wm*attn + bm (f32 [b][c][n])
  gemm_wmma_kernel<0><<<dim3(N_/128, C_/64, B_), blk, 0, stream>>>(
      Wm, bm, attn, (long)C_ * N_, nullptr, 0, C_, C_, C_, N_,
      am, nullptr, nullptr, nullptr, nullptr, nullptr, nullptr, nullptr);

  // h = relu(bn(Wc1*[desc1; am] + bc1)) — concat folded into two-source K-dim
  gemm_wmma_kernel<3><<<dim3(N_/128, C2_/64, B_), blk, 0, stream>>>(
      Wc1, bc1, desc1, (long)C_ * N_, am, (long)C_ * N_, C_, C2_, C2_, N_,
      hb, nullptr, nullptr, bng, bnb, bnm, bnv, nullptr);

  // out = desc1 + Wc2*h + bc2
  gemm_wmma_kernel<4><<<dim3(N_/128, C_/64, B_), blk, 0, stream>>>(
      Wc2, bc2, hb, (long)C2_ * N_, nullptr, 0, C2_, C_, C2_, N_,
      out, nullptr, nullptr, nullptr, nullptr, nullptr, nullptr, desc1);
}
